// Att_block_45183055954204
// MI455X (gfx1250) — matrix operand order repaired and checked by reading, not compiled, not run on hardware
//
#include <hip/hip_runtime.h>
#include <hip/hip_bf16.h>

typedef __attribute__((ext_vector_type(16))) _Float16 v16h;
typedef __attribute__((ext_vector_type(8)))  _Float16 v8h;
typedef __attribute__((ext_vector_type(8)))  float    v8f;

#define BB   16
#define TT   2048
#define CC   512
#define BT   (BB * TT)      // 32768
#define SST  2052           // padded f32 stride for S rows (breaks bank conflicts)
#define PST  2056           // padded f16 stride for P rows
#define QST  520            // padded f16 stride for staged A/Q tiles in LDS

// ---- WMMA fragment loaders (CDNA5 §7.12.2 layouts, wave32) -----------------

// A-matrix 16x32 f16: lane l -> row r=l&15; hi=l>>4 selects K halves
// elements [0..7]  = K (hi*8 .. hi*8+7), elements [8..15] = K (16+hi*8 ..)
__device__ __forceinline__ v16h frag_a_h(const _Float16* p) {
    v8h lo = *(const v8h*)p;
    v8h hi = *(const v8h*)(p + 16);
    v16h f;
#pragma unroll
    for (int i = 0; i < 8; ++i) { f[i] = lo[i]; f[i + 8] = hi[i]; }
    return f;
}

// B-matrix 32x16 f16: lane l -> column n=l&15; hi=l>>4 -> K (hi*8 .. hi*8+7), (16+hi*8 ..)
// (source is row-major [N,K], i.e. B[k][n]=M[n][k])
__device__ __forceinline__ v16h frag_b_h(const _Float16* p) {
    v8h lo = *(const v8h*)p;
    v8h hi = *(const v8h*)(p + 16);
    v16h f;
#pragma unroll
    for (int i = 0; i < 8; ++i) { f[i] = lo[i]; f[i + 8] = hi[i]; }
    return f;
}

__device__ __forceinline__ v8f wmma_f16(v16h a, v16h b, v8f c) {
    return __builtin_amdgcn_wmma_f32_16x16x32_f16(
        /*neg_a=*/false, a, /*neg_b=*/false, b,
        /*c_mod=*/(short)0, c, /*reuse_a=*/false, /*reuse_b=*/false);
}

// Async LDS-DMA: copy 64 contiguous bytes global->LDS per lane (ASYNCcnt path).
// lds_off = low 32 bits of generic shared pointer (== wave LDS byte offset).
__device__ __forceinline__ void async_copy64(unsigned lds_off, const void* gaddr) {
    asm volatile(
        "global_load_async_to_lds_b128 %0, %1, off\n\t"
        "global_load_async_to_lds_b128 %0, %1, off offset:16\n\t"
        "global_load_async_to_lds_b128 %0, %1, off offset:32\n\t"
        "global_load_async_to_lds_b128 %0, %1, off offset:48"
        :: "v"(lds_off), "v"(gaddr) : "memory");
}
__device__ __forceinline__ void wait_async0() {
    asm volatile("s_wait_asynccnt 0" ::: "memory");
}

// ---- tiny precompute: vg, sigma(qp), sigma(kp), Wq -> f16 ------------------
__global__ __launch_bounds__(512) void k_prep(
    const float* __restrict__ qp, const float* __restrict__ kp,
    const float* __restrict__ vp,
    const float* __restrict__ Wv1, const float* __restrict__ bv1,
    const float* __restrict__ Wv2, const float* __restrict__ bv2,
    const float* __restrict__ Wq,
    float* __restrict__ vg, float* __restrict__ sq, float* __restrict__ sk,
    _Float16* __restrict__ wq_h) {
    int c = threadIdx.x;           // 512 threads, one output channel each
    float d1 = 0.f, d2 = 0.f;
    for (int k = 0; k < CC; ++k) {
        float v = vp[k];
        d1 += v * Wv1[c * CC + k];
        d2 += v * Wv2[c * CC + k];
    }
    float sg = 1.f / (1.f + __expf(-(d1 + bv1[c])));
    float th = tanhf(d2 + bv2[c]);
    vg[c] = sg * th;
    sq[c] = 1.f / (1.f + __expf(-qp[c]));
    sk[c] = 1.f / (1.f + __expf(-kp[c]));
    for (int i = c; i < CC * CC; i += CC) wq_h[i] = (_Float16)Wq[i];
}

// ---- LayerNorm stage: one block per row; shared mean/var for LN1 & LN2 -----
__global__ __launch_bounds__(256) void k_ln(
    const float* __restrict__ x,
    const float* __restrict__ g1, const float* __restrict__ b1,
    const float* __restrict__ g2, const float* __restrict__ b2,
    const float* __restrict__ vg, const float* __restrict__ skg,
    _Float16* __restrict__ ln1_h, _Float16* __restrict__ k_h,
    _Float16* __restrict__ vT_h) {
    int row = blockIdx.x;               // 0..BT-1
    int bb = row >> 11, t = row & (TT - 1);
    const float* xr = x + (size_t)row * CC;
    int tid = threadIdx.x;              // 256 threads, 2 channels each
    float x0 = xr[tid * 2], x1 = xr[tid * 2 + 1];
    float s = x0 + x1, ss = x0 * x0 + x1 * x1;
#pragma unroll
    for (int m = 1; m < 32; m <<= 1) {
        s += __shfl_xor(s, m, 32);
        ss += __shfl_xor(ss, m, 32);
    }
    __shared__ float rs[8], rss[8];
    if ((tid & 31) == 0) { rs[tid >> 5] = s; rss[tid >> 5] = ss; }
    __syncthreads();
    s = 0.f; ss = 0.f;
#pragma unroll
    for (int i = 0; i < 8; ++i) { s += rs[i]; ss += rss[i]; }
    float mean = s * (1.0f / CC);
    float var  = ss * (1.0f / CC) - mean * mean;
    float rstd = rsqrtf(var + 1e-5f);
#pragma unroll
    for (int j = 0; j < 2; ++j) {
        int c = tid * 2 + j;
        float xv = (j == 0) ? x0 : x1;
        float xh = (xv - mean) * rstd;
        ln1_h[(size_t)row * CC + c] = (_Float16)(xh * g1[c] + b1[c]);
        float key = xh * g2[c] + b2[c];
        k_h[(size_t)row * CC + c] = (_Float16)(key * skg[c]);
        vT_h[((size_t)bb * CC + c) * TT + t] = (_Float16)(key * vg[c]);
    }
}

// ---- q = (LN1 @ Wq^T + bq) * sigma(qp) -------------------------------------
// One block per 16-row stripe; A tile staged via async LDS-DMA; each wave
// owns a 16x64 output slab (4 accumulators, A fragment shared across 4 Bs).
__global__ __launch_bounds__(256) void k_qgemm(
    const _Float16* __restrict__ A, const _Float16* __restrict__ W,
    const float* __restrict__ bq, const float* __restrict__ sq,
    float* __restrict__ q_f) {
    __shared__ _Float16 atile[16 * QST];   // 16.6 KB
    const int mt = blockIdx.x;             // 0..BT/16-1
    const int tid = threadIdx.x;
    {   // async stage: 16 rows x 1KB; each thread DMAs 64B of one row
        int row = tid >> 4, seg = tid & 15;
        const _Float16* ga = A + ((size_t)(mt * 16 + row)) * CC + seg * 32;
        unsigned lo = (unsigned)(uintptr_t)(atile + row * QST + seg * 32);
        async_copy64(lo, ga);
        wait_async0();
        __syncthreads();
    }
    const int w = tid >> 5, l = tid & 31, ln15 = l & 15, hi = l >> 4;
    const int c0 = w * 64;
    const _Float16* ap = atile + (size_t)ln15 * QST + hi * 8;
    v8f acc[4] = {{}, {}, {}, {}};
#pragma unroll 4
    for (int kk = 0; kk < CC; kk += 32) {
        v16h a = frag_a_h(ap + kk);
#pragma unroll
        for (int nt = 0; nt < 4; ++nt) {
            v16h bf = frag_b_h(W + (size_t)(c0 + nt * 16 + ln15) * CC + kk + hi * 8);
            acc[nt] = wmma_f16(a, bf, acc[nt]);
        }
    }
#pragma unroll
    for (int nt = 0; nt < 4; ++nt) {
        int col = c0 + nt * 16 + ln15;
        float g = sq[col], bc = bq[col];
#pragma unroll
        for (int r = 0; r < 8; ++r) {
            int row = mt * 16 + hi * 8 + r;
            q_f[(size_t)row * CC + col] = (acc[nt][r] + bc) * g;
        }
    }
}

// ---- attention: 16 query rows per WG, full score row resident in LDS -------
__global__ __launch_bounds__(256) void k_attn(
    const float* __restrict__ q_f, const _Float16* __restrict__ k_h,
    const _Float16* __restrict__ vT_h, float* __restrict__ out) {
    extern __shared__ char smem[];
    _Float16* Qh   = (_Float16*)smem;                                // 16 x QST f16
    float*    S    = (float*)(smem + 16 * QST * 2);                  // 16 x SST f32
    _Float16* P    = (_Float16*)(smem + 16 * QST * 2 + 16 * SST * 4);// 16 x PST f16
    float*    rinv = (float*)(smem + 16 * QST * 2 + 16 * SST * 4 + 16 * PST * 2);

    const int bb   = blockIdx.x >> 7;
    const int mblk = blockIdx.x & 127;
    const int tid  = threadIdx.x;
    const int w    = tid >> 5;
    const int l    = tid & 31;
    const int ln15 = l & 15;
    const int hi   = l >> 4;

    const size_t qrow0 = (size_t)bb * TT + (size_t)mblk * 16;
    const float*    qbase = q_f + qrow0 * CC;              // [16, C] f32
    const _Float16* kbase = k_h + (size_t)bb * TT * CC;    // [T, C]  f16
    const _Float16* vbase = vT_h + (size_t)bb * CC * TT;   // [C, T]  f16

    const float scale = 0.04419417382415922f;              // 1/sqrt(512)

    // Phase 0: stage Q block as f16 in LDS (convert once, reuse 16x per wave)
    {
        int row = tid >> 4, seg = tid & 15;
        const float* src = qbase + (size_t)row * CC + seg * 32;
        _Float16* dst = Qh + row * QST + seg * 32;
#pragma unroll
        for (int i = 0; i < 32; i += 8) {
            v8h h;
#pragma unroll
            for (int j = 0; j < 8; ++j) h[j] = (_Float16)src[i + j];
            *(v8h*)(dst + i) = h;
        }
    }
    __syncthreads();

    // Phase 1: S = scale * Q K^T  (each wave: 16 disjoint key tiles)
    const _Float16* aprow = Qh + (size_t)ln15 * QST + hi * 8;
    for (int i = 0; i < 16; ++i) {
        int nt = w * 16 + i;
        const _Float16* bprow = kbase + (size_t)(nt * 16 + ln15) * CC + hi * 8;
        v8f acc = {};
#pragma unroll 4
        for (int kk = 0; kk < CC; kk += 32)
            acc = wmma_f16(frag_a_h(aprow + kk), frag_b_h(bprow + kk), acc);
#pragma unroll
        for (int r = 0; r < 8; ++r)
            S[(hi * 8 + r) * SST + nt * 16 + ln15] = acc[r] * scale;
    }
    __syncthreads();

    // Phase 2: rowmax -> unnormalized exp (f16 P) -> rowsum (16 lanes per row)
    {
        int row = tid >> 4;
        int seg = tid & 15;
        float* srow = S + row * SST;
        float mx = -3.0e38f;
        for (int s = seg * 128; s < seg * 128 + 128; ++s) mx = fmaxf(mx, srow[s]);
#pragma unroll
        for (int m = 1; m < 16; m <<= 1) mx = fmaxf(mx, __shfl_xor(mx, m, 32));
        float sum = 0.f;
        _Float16* prow = P + row * PST;
        for (int s = seg * 128; s < seg * 128 + 128; ++s) {
            float e = __expf(srow[s] - mx);
            sum += e;
            prow[s] = (_Float16)e;
        }
#pragma unroll
        for (int m = 1; m < 16; m <<= 1) sum += __shfl_xor(sum, m, 32);
        if (seg == 0) rinv[row] = 1.0f / sum;
    }
    __syncthreads();

    // Phase 3: O = P @ V ; out = q + O / rowsum  (each wave owns 64 channels)
    {
        int c0 = w * 64;
        v8f acc[4] = {{}, {}, {}, {}};
        const _Float16* pap = P + (size_t)ln15 * PST + hi * 8;
        for (int s0 = 0; s0 < TT; s0 += 32) {
            v16h a = frag_a_h(pap + s0);
#pragma unroll
            for (int nt = 0; nt < 4; ++nt) {
                v16h bf = frag_b_h(vbase + (size_t)(c0 + nt * 16 + ln15) * TT + s0 + hi * 8);
                acc[nt] = wmma_f16(a, bf, acc[nt]);
            }
        }
#pragma unroll
        for (int nt = 0; nt < 4; ++nt) {
            int col = c0 + nt * 16 + ln15;
#pragma unroll
            for (int r = 0; r < 8; ++r) {
                int m = hi * 8 + r;
                size_t idx = (qrow0 + m) * CC + col;
                out[idx] = q_f[idx] + acc[nt][r] * rinv[m];
            }
        }
    }
}

// ---------------------------------------------------------------------------
extern "C" void kernel_launch(void* const* d_in, const int* in_sizes, int n_in,
                              void* d_out, int out_size, void* d_ws, size_t ws_size,
                              hipStream_t stream) {
    (void)in_sizes; (void)n_in; (void)out_size; (void)ws_size;
    const float* x   = (const float*)d_in[0];
    const float* g1  = (const float*)d_in[1];
    const float* b1  = (const float*)d_in[2];
    const float* g2  = (const float*)d_in[3];
    const float* b2  = (const float*)d_in[4];
    const float* qp  = (const float*)d_in[5];
    const float* kp  = (const float*)d_in[6];
    const float* vp  = (const float*)d_in[7];
    const float* Wq  = (const float*)d_in[8];
    const float* bq  = (const float*)d_in[9];
    const float* Wv1 = (const float*)d_in[10];
    const float* bv1 = (const float*)d_in[11];
    const float* Wv2 = (const float*)d_in[12];
    const float* bv2 = (const float*)d_in[13];
    float* out = (float*)d_out;

    char* ws = (char*)d_ws;
    size_t off = 0;
    auto take = [&](size_t bytes) -> char* {
        char* p = ws + off;
        off += (bytes + 255) & ~(size_t)255;
        return p;
    };
    _Float16* ln1_h = (_Float16*)take((size_t)BT * CC * 2);  // LN1 output (f16)
    _Float16* k_h   = (_Float16*)take((size_t)BT * CC * 2);  // gated K (f16)
    _Float16* vT_h  = (_Float16*)take((size_t)BT * CC * 2);  // gated V transposed (f16)
    _Float16* wq_h  = (_Float16*)take((size_t)CC * CC * 2);  // Wq (f16)
    float*    q_f   = (float*)take((size_t)BT * CC * 4);     // gated q (f32)
    float*    vg    = (float*)take(CC * 4);
    float*    sq    = (float*)take(CC * 4);
    float*    sk    = (float*)take(CC * 4);

    hipLaunchKernelGGL(k_prep, dim3(1), dim3(512), 0, stream,
                       qp, kp, vp, Wv1, bv1, Wv2, bv2, Wq, vg, sq, sk, wq_h);
    hipLaunchKernelGGL(k_ln, dim3(BT), dim3(256), 0, stream,
                       x, g1, b1, g2, b2, vg, sk, ln1_h, k_h, vT_h);
    hipLaunchKernelGGL(k_qgemm, dim3(BT / 16), dim3(256), 0, stream,
                       ln1_h, wq_h, bq, sq, q_f);
    size_t lds = (size_t)16 * QST * 2 + (size_t)16 * SST * 4 + (size_t)16 * PST * 2 + 64;
    hipLaunchKernelGGL(k_attn, dim3(BB * (TT / 16)), dim3(256), lds, stream,
                       q_f, k_h, vT_h, out);
}
